// CustomRetrieverModel_23149873725659
// MI455X (gfx1250) — matrix operand order repaired and checked by reading, not compiled, not run on hardware
//
#include <hip/hip_runtime.h>
#include <hip/hip_bf16.h>
#include <stdint.h>

#define HD 768
#define NB 32
#define LQ 64
#define NM 64
#define LDT 256
#define NL 3
#define BPG 4                        // batches per workgroup

// LDS staging: rows of 64B data padded to 80B stride (TDM pad: 16 DW + 4 DW)
#define ROWSTRIDE 80
#define LDS_T_SZ (256 * ROWSTRIDE)   // 20480 B per buffer (256 rows x 32 K bf16)

typedef __attribute__((ext_vector_type(16))) __bf16 v16bf;
typedef __attribute__((ext_vector_type(8)))  float  v8f;
typedef __attribute__((ext_vector_type(4)))  unsigned int u32x4;
typedef __attribute__((ext_vector_type(8)))  int  i32x8;
typedef __attribute__((ext_vector_type(4)))  int  i32x4;

union FragU { v16bf v; uint4 q[2]; };

__device__ __forceinline__ uint16_t f32_to_bf16_rne(float f) {
  uint32_t u = __float_as_uint(f);
  u += 0x7FFFu + ((u >> 16) & 1u);
  return (uint16_t)(u >> 16);
}

// TDM D# group1: 2D bf16 tile of (256 x 32) from a 768-stride tensor,
// LDS padding 16DW->+4DW (80B row stride in LDS, bank-conflict-free frags).
__device__ __forceinline__ i32x8 make_g1() {
  i32x8 g;
  g[0] = (1 << 16)          // data_size = 2B
       | (1 << 20)          // pad_enable
       | (3 << 22)          // pad_interval: 16 DWORDs
       | (3 << 25);         // pad_amount: 4 DWORDs
  g[1] = (HD & 0xFFFF) << 16;                   // tensor_dim0 lo = 768
  g[2] = (HD >> 16) | ((256 & 0xFFFF) << 16);   // tensor_dim0 hi | tensor_dim1 lo
  g[3] = 0 | (32 << 16);                        // tensor_dim1 hi | tile_dim0 = 32
  g[4] = 256;                                   // tile_dim1 = 256, tile_dim2 = 0
  g[5] = HD;                                    // tensor_dim0_stride = 768
  g[6] = 0;
  g[7] = 0;
  return g;
}

__device__ __forceinline__ void tdm_issue(uint32_t lds_off, uint64_t gaddr,
                                          i32x8 g1) {
  u32x4 g0;
  g0[0] = 1u;                                   // count=1, user mode
  g0[1] = lds_off;                              // lds_addr
  g0[2] = (uint32_t)gaddr;                      // global_addr[31:0]
  g0[3] = ((uint32_t)(gaddr >> 32) & 0x1FFFFFFu) | (2u << 30); // addr hi | type=2
  i32x4 z4 = {0, 0, 0, 0};
  i32x8 z8 = {0, 0, 0, 0, 0, 0, 0, 0};
  __builtin_amdgcn_tensor_load_to_lds(g0, g1, z4, z4, z8, 0);
}

// ---------------------------------------------------------------------------
// Phase 1a: per-row L2 normalize (+ optional punct mask), f32 -> bf16
// ---------------------------------------------------------------------------
__global__ __launch_bounds__(256)
void norm_rows_bf16_k(const float* __restrict__ in,
                      const unsigned char* __restrict__ mask,
                      uint16_t* __restrict__ out) {
  const int row = blockIdx.x;
  const float* src = in + (size_t)row * HD;
  float mval = 1.0f;
  if (mask) mval = mask[row] ? 1.0f : 0.0f;
  float v0 = src[threadIdx.x]       * mval;
  float v1 = src[threadIdx.x + 256] * mval;
  float v2 = src[threadIdx.x + 512] * mval;
  float ss = v0 * v0 + v1 * v1 + v2 * v2;
#pragma unroll
  for (int s = 1; s < 32; s <<= 1) ss += __shfl_xor(ss, s, 32);
  __shared__ float red[8];
  __shared__ float scale_sh;
  if ((threadIdx.x & 31) == 0) red[threadIdx.x >> 5] = ss;
  __syncthreads();
  if (threadIdx.x == 0) {
    float t = 0.0f;
#pragma unroll
    for (int i = 0; i < 8; ++i) t += red[i];
    scale_sh = 1.0f / fmaxf(sqrtf(t), 1e-12f);
  }
  __syncthreads();
  const float sc = scale_sh;
  uint16_t* dst = out + (size_t)row * HD;
  dst[threadIdx.x]       = f32_to_bf16_rne(v0 * sc);
  dst[threadIdx.x + 256] = f32_to_bf16_rne(v1 * sc);
  dst[threadIdx.x + 512] = f32_to_bf16_rne(v2 * sc);
}

// ---------------------------------------------------------------------------
// Phase 1b: per-row L2 normalize, f32 -> f32 (CLS vectors)
// ---------------------------------------------------------------------------
__global__ __launch_bounds__(256)
void norm_rows_f32_k(const float* __restrict__ in, float* __restrict__ out) {
  const int row = blockIdx.x;
  const float* src = in + (size_t)row * HD;
  float v0 = src[threadIdx.x];
  float v1 = src[threadIdx.x + 256];
  float v2 = src[threadIdx.x + 512];
  float ss = v0 * v0 + v1 * v1 + v2 * v2;
#pragma unroll
  for (int s = 1; s < 32; s <<= 1) ss += __shfl_xor(ss, s, 32);
  __shared__ float red[8];
  __shared__ float scale_sh;
  if ((threadIdx.x & 31) == 0) red[threadIdx.x >> 5] = ss;
  __syncthreads();
  if (threadIdx.x == 0) {
    float t = 0.0f;
#pragma unroll
    for (int i = 0; i < 8; ++i) t += red[i];
    scale_sh = 1.0f / fmaxf(sqrtf(t), 1e-12f);
  }
  __syncthreads();
  const float sc = scale_sh;
  float* dst = out + (size_t)row * HD;
  dst[threadIdx.x]       = v0 * sc;
  dst[threadIdx.x + 256] = v1 * sc;
  dst[threadIdx.x + 512] = v2 * sc;
}

// ---------------------------------------------------------------------------
// Phase 2: TDM double-buffered LDS staging + WMMA + fused masked max/avg.
// One workgroup (16 waves) per (4 batches, m). Wave w: b_local = w>>2,
// Lq tile = w&3. Per K-chunk (32): wave0 DMAs next doc chunk (256x32) and
// q chunk (4 batches x 64 rows = 256x32) while all waves WMMA from LDS.
// Doc-chunk DMA is amortized over 4 batches.
// ---------------------------------------------------------------------------
__global__ __launch_bounds__(512)
void sim_maxavg_k(const uint16_t* __restrict__ qn,   // (B,Lq,H) bf16 normed
                  const uint16_t* __restrict__ dn,   // (M,Ld,H) bf16 normed
                  const int* __restrict__ q_ids,     // (B,Lq)
                  const int* __restrict__ d_ids,     // (M,Ld)
                  float* __restrict__ avg_sim) {     // (B,M)
  __shared__ __align__(128) unsigned char lds_d[2][LDS_T_SZ];
  __shared__ __align__(128) unsigned char lds_a[2][LDS_T_SZ];

  const int bgrp = blockIdx.x >> 6;          // 0..NB/BPG-1
  const int m    = blockIdx.x & 63;
  const int wave = threadIdx.x >> 5;         // 0..15
  const int bl   = wave >> 2;                // batch within group
  const int qt   = wave & 3;                 // Lq tile
  const int lane = threadIdx.x & 31;
  const int r = lane & 15;
  const int h = lane >> 4;
  const int b = bgrp * BPG + bl;

  const uint64_t gaD = (uint64_t)(uintptr_t)(dn + (size_t)m * LDT * HD);
  const uint64_t gaA = (uint64_t)(uintptr_t)(qn + (size_t)bgrp * BPG * LQ * HD);
  const i32x8 g1 = make_g1();
  const uint32_t ldsoff_d0 = (uint32_t)(uintptr_t)&lds_d[0][0];
  const uint32_t ldsoff_d1 = (uint32_t)(uintptr_t)&lds_d[1][0];
  const uint32_t ldsoff_a0 = (uint32_t)(uintptr_t)&lds_a[0][0];
  const uint32_t ldsoff_a1 = (uint32_t)(uintptr_t)&lds_a[1][0];

  v8f acc[16];
  const v8f zero = {0.f, 0.f, 0.f, 0.f, 0.f, 0.f, 0.f, 0.f};
#pragma unroll
  for (int dt = 0; dt < 16; ++dt) acc[dt] = zero;

  // Prologue: DMA chunk 0 into buffer 0.
  if (wave == 0) {
    tdm_issue(ldsoff_d0, gaD, g1);
    tdm_issue(ldsoff_a0, gaA, g1);
    __builtin_amdgcn_s_wait_tensorcnt(0);
  }
  __syncthreads();

  int cur = 0;
  for (int k0 = 0; k0 < HD; k0 += 32) {
    const int nxt = cur ^ 1;
    const bool more = (k0 + 32) < HD;
    if (wave == 0 && more) {
      tdm_issue(nxt ? ldsoff_d1 : ldsoff_d0, gaD + (size_t)(k0 + 32) * 2, g1);
      tdm_issue(nxt ? ldsoff_a1 : ldsoff_a0, gaA + (size_t)(k0 + 32) * 2, g1);
    }

    // A fragment: lane holds q row (bl*64 + qt*16 + r) within the staged tile;
    // contiguous K half {16h..16h+15} (matches B).
    const unsigned char* abase =
        &lds_a[cur][(size_t)(bl * 64 + qt * 16 + r) * ROWSTRIDE];
    FragU a;
    a.q[0] = *reinterpret_cast<const uint4*>(abase + 32 * h);
    a.q[1] = *reinterpret_cast<const uint4*>(abase + 32 * h + 16);

    // B fragments: lane holds doc column (dt*16+r), contiguous K half h*16.
    const unsigned char* dbase = &lds_d[cur][(size_t)r * ROWSTRIDE + h * 32];
    FragU bfr[2];
    bfr[0].q[0] = *reinterpret_cast<const uint4*>(dbase);
    bfr[0].q[1] = *reinterpret_cast<const uint4*>(dbase + 16);
#pragma unroll
    for (int dt = 0; dt < 16; ++dt) {
      if (dt < 15) {  // ping-pong prefetch of next fragment from LDS
        bfr[(dt + 1) & 1].q[0] =
            *reinterpret_cast<const uint4*>(dbase + (size_t)(dt + 1) * 16 * ROWSTRIDE);
        bfr[(dt + 1) & 1].q[1] =
            *reinterpret_cast<const uint4*>(dbase + (size_t)(dt + 1) * 16 * ROWSTRIDE + 16);
      }
      acc[dt] = __builtin_amdgcn_wmma_f32_16x16x32_bf16(
          false, a.v, false, bfr[dt & 1].v, (short)0, acc[dt], false, false);
    }

    if (wave == 0 && more) __builtin_amdgcn_s_wait_tensorcnt(0);
    __syncthreads();
    cur = nxt;
  }

  // Masked running max over doc tokens. C layout: lane col = r, VGPR v -> row v+8h.
  float rmax[8];
#pragma unroll
  for (int v = 0; v < 8; ++v) rmax[v] = -1e30f;
#pragma unroll
  for (int dt = 0; dt < 16; ++dt) {
    const bool dvalid = d_ids[m * LDT + dt * 16 + r] != 0;
#pragma unroll
    for (int v = 0; v < 8; ++v) {
      float x = dvalid ? acc[dt][v] : -1e-9f;   // MASK_FILL
      rmax[v] = fmaxf(rmax[v], x);
    }
  }
#pragma unroll
  for (int s = 1; s < 16; s <<= 1) {
#pragma unroll
    for (int v = 0; v < 8; ++v)
      rmax[v] = fmaxf(rmax[v], __shfl_xor(rmax[v], s, 32));
  }
  float sum = 0.0f;
#pragma unroll
  for (int v = 0; v < 8; ++v) {
    const int qr = qt * 16 + v + 8 * h;
    const float w = (q_ids[b * LQ + qr] != 0) ? 1.0f : 0.0f;
    sum += rmax[v] * w;
  }
  sum += __shfl_xor(sum, 16, 32);

  __shared__ float wsum[16];
  if (lane == 0) wsum[wave] = sum;
  __syncthreads();
  if (threadIdx.x < BPG) {
    const int t = threadIdx.x;
    float tot = wsum[t * 4] + wsum[t * 4 + 1] + wsum[t * 4 + 2] + wsum[t * 4 + 3];
    const int bb = bgrp * BPG + t;
    int nv = 0;
    for (int i = 0; i < LQ; ++i) nv += (q_ids[bb * LQ + i] != 0) ? 1 : 0;
    avg_sim[bb * NM + m] = tot / ((float)nv + 1e-10f);
  }
}

// ---------------------------------------------------------------------------
// Phase 3: Wq (CLS layer weighting) + even/odd shuffle + temperature scale.
// ---------------------------------------------------------------------------
__global__ __launch_bounds__(128)
void wq_final_k(const float* __restrict__ qcn,      // (L,B,H) normed
                const float* __restrict__ dcn,      // (L,M,H) normed
                const float* __restrict__ avg_sim,  // (B,M)
                const float* __restrict__ log_inv_t,
                float* __restrict__ out) {          // (B,M)
  const int pair = blockIdx.x * 4 + (threadIdx.x >> 5);
  const int lane = threadIdx.x & 31;
  const int b = pair >> 6;
  const int m = pair & 63;
  const float* q   = qcn + ((size_t)((NL - 1) * NB + b)) * HD;
  const float* d0p = dcn + ((size_t)(0 * NM + m)) * HD;
  const float* d1p = dcn + ((size_t)(1 * NM + m)) * HD;
  const float* d2p = dcn + ((size_t)(2 * NM + m)) * HD;
  float d0 = 0.f, d1 = 0.f, d2 = 0.f;
  for (int i = lane; i < HD; i += 32) {
    const float qv = q[i];
    d0 += qv * d0p[i];
    d1 += qv * d1p[i];
    d2 += qv * d2p[i];
  }
#pragma unroll
  for (int s = 1; s < 32; s <<= 1) {
    d0 += __shfl_xor(d0, s, 32);
    d1 += __shfl_xor(d1, s, 32);
    d2 += __shfl_xor(d2, s, 32);
  }
  if (lane == 0) {
    const float wq = (d2 - fminf(d0, d1)) * 0.5f;
    const float invt = expf(log_inv_t[0]);
    const float val = avg_sim[b * NM + m] * wq * invt;
    const int j = (m & 1) ? (NM / 2 + (m >> 1)) : (m >> 1);
    out[b * NM + j] = val;
  }
}

// ---------------------------------------------------------------------------
extern "C" void kernel_launch(void* const* d_in, const int* in_sizes, int n_in,
                              void* d_out, int out_size, void* d_ws, size_t ws_size,
                              hipStream_t stream) {
  const float* q_tok    = (const float*)d_in[0];
  const float* d_tok    = (const float*)d_in[1];
  const float* q_cls    = (const float*)d_in[2];
  const float* d_cls    = (const float*)d_in[3];
  const float* log_invt = (const float*)d_in[4];
  const int*   q_ids    = (const int*)d_in[5];
  const int*   d_ids    = (const int*)d_in[6];
  const unsigned char* d_punct = (const unsigned char*)d_in[7];

  char* ws = (char*)d_ws;
  uint16_t* qn  = (uint16_t*)(ws + 0);          //  3,145,728 B
  uint16_t* dn  = (uint16_t*)(ws + 3145728);    // 25,165,824 B
  float*    qcn = (float*)   (ws + 28311552);   //    294,912 B
  float*    dcn = (float*)   (ws + 28606464);   //    589,824 B
  float*    avg = (float*)   (ws + 29196288);   //      8,192 B

  norm_rows_bf16_k<<<NB * LQ, 256, 0, stream>>>(q_tok, nullptr, qn);
  norm_rows_bf16_k<<<NM * LDT, 256, 0, stream>>>(d_tok, d_punct, dn);
  norm_rows_f32_k<<<NL * NB, 256, 0, stream>>>(q_cls, qcn);
  norm_rows_f32_k<<<NL * NM, 256, 0, stream>>>(d_cls, dcn);

  sim_maxavg_k<<<(NB / BPG) * NM, 512, 0, stream>>>(qn, dn, q_ids, d_ids, avg);

  wq_final_k<<<(NB * NM) / 4, 128, 0, stream>>>(qcn, dcn, avg, log_invt,
                                                (float*)d_out);
}
